// FMoE_88407606821256
// MI455X (gfx1250) — matrix operand order repaired and checked by reading, not compiled, not run on hardware
//
#include <hip/hip_runtime.h>
#include <hip/hip_bf16.h>
#include <math.h>

// MoE: T=8192 tokens, D=512, H=1024, E=16 experts, top-2 routing.
#define NTOK 8192
#define DM   512
#define DH   1024
#define NE   16
#define MB   32            // tokens per block (2 WMMA M-tiles) -> 2x B reuse

typedef __bf16 bf16;
typedef __attribute__((ext_vector_type(16))) __bf16 v16bf;
typedef __attribute__((ext_vector_type(8)))  float  v8f;

union ABFrag { v16bf v; uint4 q[2]; };

// ---------------------------------------------------------------------------
// WMMA fragment loaders (v_wmma_f32_16x16x32_bf16 operands, wave32; A/B in this code's contiguous K order)
// A (16x32, row-major in LDS): lane holds row m=lane&15;
//   elems 0..7  -> K = kb+0..7   (kb = k0 + (lane>=16 ? 16 : 0))
//   elems 8..15 -> K = kb+8..15         => two 16B LDS reads
// B (32x16, K-contiguous in memory): lane holds col n=n0+(lane&15);
//   elems 0..15 -> K = kb+0..15  (kb = k0 + (lane>=16 ? 16 : 0)) => 32B read
// C/D (16x16 f32): elem r -> row M = r + (lane>=16 ? 8 : 0), col N = lane&15
// ---------------------------------------------------------------------------
__device__ __forceinline__ v16bf loadA_lds(const bf16* base, int ldk, int k0, int lane) {
  int m  = lane & 15;
  int kb = k0 + ((lane & 16) ? 16 : 0);
  ABFrag u;
  u.q[0] = *(const uint4*)(base + m * ldk + kb);
  u.q[1] = *(const uint4*)(base + m * ldk + kb + 8);
  return u.v;
}

__device__ __forceinline__ v16bf loadB_glb(const bf16* __restrict__ W, int ldk,
                                           int n0, int k0, int lane) {
  int n  = n0 + (lane & 15);
  int kb = k0 + ((lane & 16) ? 16 : 0);
  const uint4* p = (const uint4*)(W + (size_t)n * ldk + kb);
  ABFrag u;
  u.q[0] = p[0];
  u.q[1] = p[1];
  __builtin_prefetch((const void*)(p + 4), 0, 0);  // next K-step (64B ahead)
  return u.v;
}

__device__ __forceinline__ float gelu_exact(float v) {
  return 0.5f * v * (1.0f + erff(v * 0.70710678118654752f));
}

// ---------------------------------------------------------------------------
// One-time fp32 -> bf16 converts. Weights transposed so the WMMA B operand is
// K-contiguous: W1 [E][D][H] -> W1t [E][H][D]; W2 [E][H][D] -> W2t [E][D][H].
// x -> xbf (row-major) so the FFN can async-gather rows straight into LDS.
// ---------------------------------------------------------------------------
__global__ __launch_bounds__(256) void conv_x_kernel(const float* __restrict__ x,
                                                     bf16* __restrict__ xbf) {
  size_t i = (size_t)blockIdx.x * 256 + threadIdx.x;
  if (i < (size_t)NTOK * DM) xbf[i] = (bf16)x[i];
}

__global__ __launch_bounds__(256) void conv_w1_kernel(const float* __restrict__ W1,
                                                      bf16* __restrict__ W1t) {
  size_t i = (size_t)blockIdx.x * 256 + threadIdx.x;
  if (i >= (size_t)NE * DM * DH) return;
  int e   = (int)(i / (DM * DH));
  int rem = (int)(i % (DM * DH));
  int d   = rem / DH;
  int h   = rem % DH;
  W1t[(size_t)e * DH * DM + (size_t)h * DM + d] = (bf16)W1[i];
}

__global__ __launch_bounds__(256) void conv_w2_kernel(const float* __restrict__ W2,
                                                      bf16* __restrict__ W2t) {
  size_t i = (size_t)blockIdx.x * 256 + threadIdx.x;
  if (i >= (size_t)NE * DH * DM) return;
  int e   = (int)(i / (DH * DM));
  int rem = (int)(i % (DH * DM));
  int h   = rem / DM;
  int d   = rem % DM;
  W2t[(size_t)e * DM * DH + (size_t)d * DH + h] = (bf16)W2[i];
}

// ---------------------------------------------------------------------------
// Gate: logits = x@Wg + bg, top-2, softmax over the 2, bucketize per expert.
// ---------------------------------------------------------------------------
__global__ __launch_bounds__(256) void gate_kernel(const float* __restrict__ x,
                                                   const float* __restrict__ Wg,
                                                   const float* __restrict__ bg,
                                                   int* __restrict__ cnt,
                                                   int* __restrict__ btok,
                                                   float* __restrict__ bw,
                                                   int* __restrict__ bslot) {
  __shared__ float wgl[DM * NE];
  int tid = threadIdx.x;
  for (int i = tid; i < DM * NE; i += 256) wgl[i] = Wg[i];
  __syncthreads();

  int t = blockIdx.x * 256 + tid;
  if (t >= NTOK) return;

  float acc[NE];
#pragma unroll
  for (int e = 0; e < NE; ++e) acc[e] = bg[e];

  const float* xr = x + (size_t)t * DM;
  for (int d = 0; d < DM; ++d) {
    float xv = xr[d];
#pragma unroll
    for (int e = 0; e < NE; ++e) acc[e] = fmaf(xv, wgl[d * NE + e], acc[e]);
  }

  float best = -1e30f, sec = -1e30f;
  int bi = 0, si = 0;
#pragma unroll
  for (int e = 0; e < NE; ++e) {
    float v = acc[e];
    if (v > best) { sec = best; si = bi; best = v; bi = e; }
    else if (v > sec) { sec = v; si = e; }
  }
  float e1 = __expf(sec - best);
  float s  = 1.0f + e1;
  float w0 = 1.0f / s;
  float w1 = e1 / s;

  int p0 = atomicAdd(&cnt[bi], 1);
  btok[bi * NTOK + p0] = t; bw[bi * NTOK + p0] = w0; bslot[bi * NTOK + p0] = 0;
  int p1 = atomicAdd(&cnt[si], 1);
  btok[si * NTOK + p1] = t; bw[si * NTOK + p1] = w1; bslot[si * NTOK + p1] = 1;
}

// ---------------------------------------------------------------------------
// Fused per-expert FFN, MB=32 tokens/block (2 M-tiles), 8 waves.
// Async-gather x rows (bf16) -> LDS, GEMM1(K=512)+GELU -> LDS, GEMM2(K=1024),
// scaled write into per-slot partial buffer (deterministic, no float atomics).
// Each B fragment is reused across both M-tiles (halves L2 B traffic/FLOP).
// Dynamic LDS layout (base = LDS offset 0, no static LDS in this kernel):
//   [0, 32KB)        xt : bf16 [MB][DM]
//   [32KB, 96KB)     ht : bf16 [MB][DH]
//   [96KB, ...)      toks/wts/slots/rvalid : int/float [MB]
// ---------------------------------------------------------------------------
#define XT_BYTES (MB * DM * 2)
#define HT_BYTES (MB * DH * 2)
#define META_OFF (XT_BYTES + HT_BYTES)
#define SMEM_BYTES (META_OFF + MB * 16)

__global__ __launch_bounds__(256) void ffn_kernel(const bf16*  __restrict__ xbf,
                                                  const bf16*  __restrict__ W1t,
                                                  const bf16*  __restrict__ W2t,
                                                  const float* __restrict__ b1,
                                                  const float* __restrict__ b2,
                                                  const int*   __restrict__ cnt,
                                                  const int*   __restrict__ btok,
                                                  const float* __restrict__ bw,
                                                  const int*   __restrict__ bslot,
                                                  float* __restrict__ part) {
  extern __shared__ __attribute__((aligned(16))) char smem[];
  bf16*  xt     = (bf16*)smem;
  bf16*  ht     = (bf16*)(smem + XT_BYTES);
  int*   toks   = (int*)(smem + META_OFF);
  float* wts    = (float*)(smem + META_OFF + MB * 4);
  int*   slots  = (int*)(smem + META_OFF + MB * 8);
  int*   rvalid = (int*)(smem + META_OFF + MB * 12);

  int e    = blockIdx.y;
  int tile = blockIdx.x;
  int n    = cnt[e];
  if (tile * MB >= n) return;

  int tid = threadIdx.x;
  if (tid < MB) {
    int row  = tile * MB + tid;
    int ok   = row < n;
    toks[tid]   = ok ? btok[e * NTOK + row] : 0;
    wts[tid]    = ok ? bw[e * NTOK + row] : 0.0f;
    slots[tid]  = ok ? bslot[e * NTOK + row] : 0;
    rvalid[tid] = ok;
  }
  __syncthreads();

  // Async-gather routed bf16 token rows into LDS (16B chunks, ASYNCcnt path).
  // xt is at LDS byte offset 0 and row-major, so dest offset == i*16.
  {
    const int CH = DM / 8;                       // 64 16B-chunks per row
    for (int i = tid; i < MB * CH; i += 256) {
      int r = i >> 6;
      int c = i & (CH - 1);
      const bf16* src = xbf + (size_t)toks[r] * DM + c * 8;
      unsigned dstoff = (unsigned)(i * 16);
      asm volatile("global_load_async_to_lds_b128 %0, %1, off"
                   :: "v"(dstoff), "v"(src) : "memory");
    }
    asm volatile("s_wait_asynccnt 0x0" ::: "memory");
  }
  __syncthreads();

  int lane = tid & 31;
  int wave = tid >> 5;
  int mb   = (lane & 16) ? 8 : 0;
  int nlo  = lane & 15;

  // ---- GEMM1: h = gelu(x @ W1[e] + b1[e]) ----
  const bf16* W1e = W1t + (size_t)e * DH * DM;
#pragma unroll
  for (int j = 0; j < DH / (16 * 8); ++j) {      // 8 N-tiles per wave
    int n0 = (wave + j * 8) * 16;
    v8f acc0 = {0.f, 0.f, 0.f, 0.f, 0.f, 0.f, 0.f, 0.f};
    v8f acc1 = {0.f, 0.f, 0.f, 0.f, 0.f, 0.f, 0.f, 0.f};
    for (int k0 = 0; k0 < DM; k0 += 32) {
      v16bf b  = loadB_glb(W1e, DM, n0, k0, lane);   // loaded once,
      v16bf a0 = loadA_lds(xt,            DM, k0, lane);
      v16bf a1 = loadA_lds(xt + 16 * DM,  DM, k0, lane);
      acc0 = __builtin_amdgcn_wmma_f32_16x16x32_bf16(false, a0, false, b,
                                                     (short)0, acc0, false, false);
      acc1 = __builtin_amdgcn_wmma_f32_16x16x32_bf16(false, a1, false, b,
                                                     (short)0, acc1, false, false);
    }
    int   nc   = n0 + nlo;
    float bias = b1[e * DH + nc];
#pragma unroll
    for (int r = 0; r < 8; ++r) {
      float v0 = acc0[r] + bias;
      float v1 = acc1[r] + bias;
      ht[(mb + r) * DH + nc]        = (bf16)gelu_exact(v0);
      ht[(16 + mb + r) * DH + nc]   = (bf16)gelu_exact(v1);
    }
  }
  __syncthreads();

  // ---- GEMM2: y = h @ W2[e] + b2[e]; write w * y into per-slot partial ----
  const bf16* W2e = W2t + (size_t)e * DM * DH;
#pragma unroll
  for (int j = 0; j < DM / (16 * 8); ++j) {      // 4 N-tiles per wave
    int n0 = (wave + j * 8) * 16;
    v8f acc0 = {0.f, 0.f, 0.f, 0.f, 0.f, 0.f, 0.f, 0.f};
    v8f acc1 = {0.f, 0.f, 0.f, 0.f, 0.f, 0.f, 0.f, 0.f};
    for (int k0 = 0; k0 < DH; k0 += 32) {
      v16bf b  = loadB_glb(W2e, DH, n0, k0, lane);
      v16bf a0 = loadA_lds(ht,            DH, k0, lane);
      v16bf a1 = loadA_lds(ht + 16 * DH,  DH, k0, lane);
      acc0 = __builtin_amdgcn_wmma_f32_16x16x32_bf16(false, a0, false, b,
                                                     (short)0, acc0, false, false);
      acc1 = __builtin_amdgcn_wmma_f32_16x16x32_bf16(false, a1, false, b,
                                                     (short)0, acc1, false, false);
    }
    int   nc   = n0 + nlo;
    float bias = b2[e * DM + nc];
#pragma unroll
    for (int mi = 0; mi < 2; ++mi) {
#pragma unroll
      for (int r = 0; r < 8; ++r) {
        int m = mi * 16 + mb + r;
        if (rvalid[m]) {
          float a = (mi == 0) ? acc0[r] : acc1[r];
          float v = wts[m] * (a + bias);
          part[(size_t)slots[m] * NTOK * DM + (size_t)toks[m] * DM + nc] = v;
        }
      }
    }
  }
}

// out = part[slot0] + part[slot1]  (each (slot,token,d) written exactly once)
__global__ __launch_bounds__(256) void combine_kernel(const float* __restrict__ part,
                                                      float* __restrict__ out) {
  size_t i = ((size_t)blockIdx.x * 256 + threadIdx.x) * 4;
  if (i >= (size_t)NTOK * DM) return;
  const float4 a = *(const float4*)(part + i);
  const float4 b = *(const float4*)(part + (size_t)NTOK * DM + i);
  float4 o;
  o.x = a.x + b.x; o.y = a.y + b.y; o.z = a.z + b.z; o.w = a.w + b.w;
  *(float4*)(out + i) = o;
}

// ---------------------------------------------------------------------------
extern "C" void kernel_launch(void* const* d_in, const int* in_sizes, int n_in,
                              void* d_out, int out_size, void* d_ws, size_t ws_size,
                              hipStream_t stream) {
  const float* x  = (const float*)d_in[0];
  const float* Wg = (const float*)d_in[1];
  const float* bg = (const float*)d_in[2];
  const float* W1 = (const float*)d_in[3];
  const float* b1 = (const float*)d_in[4];
  const float* W2 = (const float*)d_in[5];
  const float* b2 = (const float*)d_in[6];
  float* out = (float*)d_out;

  // Workspace layout
  char* ws = (char*)d_ws;
  const size_t W_BYTES = (size_t)NE * DM * DH * sizeof(bf16);   // 16 MB each
  const size_t X_BYTES = (size_t)NTOK * DM * sizeof(bf16);      // 8 MB
  const size_t P_BYTES = (size_t)2 * NTOK * DM * sizeof(float); // 32 MB
  bf16*  W1t   = (bf16*)(ws);
  bf16*  W2t   = (bf16*)(ws + W_BYTES);
  bf16*  xbf   = (bf16*)(ws + 2 * W_BYTES);
  float* part  = (float*)(ws + 2 * W_BYTES + X_BYTES);
  char*  tail  = ws + 2 * W_BYTES + X_BYTES + P_BYTES;
  int*   cnt   = (int*)(tail);
  int*   btok  = (int*)(tail + 256);
  float* bwgt  = (float*)(tail + 256 + (size_t)NE * NTOK * 4);
  int*   bslot = (int*)(tail + 256 + (size_t)NE * NTOK * 8);

  hipMemsetAsync(cnt, 0, NE * sizeof(int), stream);

  int xElems = NTOK * DM;
  int wElems = NE * DM * DH;
  conv_x_kernel<<<(xElems + 255) / 256, 256, 0, stream>>>(x, xbf);
  conv_w1_kernel<<<(wElems + 255) / 256, 256, 0, stream>>>(W1, W1t);
  conv_w2_kernel<<<(wElems + 255) / 256, 256, 0, stream>>>(W2, W2t);

  gate_kernel<<<NTOK / 256, 256, 0, stream>>>(x, Wg, bg, cnt, btok, bwgt, bslot);

  ffn_kernel<<<dim3(NTOK / MB, NE), 256, SMEM_BYTES, stream>>>(
      xbf, W1t, W2t, b1, b2, cnt, btok, bwgt, bslot, part);

  combine_kernel<<<(NTOK * DM / 4 + 255) / 256, 256, 0, stream>>>(part, out);
}
